// EEGGCNLayer_53094385713628
// MI455X (gfx1250) — matrix operand order repaired and checked by reading, not compiled, not run on hardware
//
#include <hip/hip_runtime.h>
#include <hip/hip_bf16.h>

// EEG GCN layer for MI455X (gfx1250, wave32).
// Pipeline: prep(cast->bf16, zero acc, deg=1) -> degree atomics ->
//           bf16 WMMA GEMM (v_wmma_f32_16x16x32_bf16) -> edge scatter atomics ->
//           fused bias + self-loop + LayerNorm + ReLU.

typedef __bf16 bf16;
typedef __attribute__((ext_vector_type(16))) __bf16 v16bf;
typedef __attribute__((ext_vector_type(8)))  float  v8f;

#define IN_CH  128
#define OUT_CH 128
#define LN_EPS 1e-5f

// ---------------------------------------------------------------------------
// prep: xb = bf16(x); wbt[n][k] = bf16(W[k][n]); acc = 0; deg = 1 (self loop)
// ---------------------------------------------------------------------------
__global__ void gcn_prep(const float* __restrict__ x, const float* __restrict__ W,
                         bf16* __restrict__ xb, bf16* __restrict__ wbt,
                         float* __restrict__ deg, float* __restrict__ acc,
                         int total, int nNodes)
{
    int i = blockIdx.x * blockDim.x + threadIdx.x;
    if (i >= total) return;
    xb[i]  = (bf16)x[i];
    acc[i] = 0.0f;
    if (i < nNodes) deg[i] = 1.0f;               // self-loop contribution to degree
    if (i < IN_CH * OUT_CH) {
        int n = i >> 7;                           // output channel
        int k = i & (IN_CH - 1);                  // input channel
        wbt[i] = (bf16)W[k * OUT_CH + n];         // transposed copy: wbt[n][k]
    }
}

// ---------------------------------------------------------------------------
// degree: deg[col[e]] += 1
// ---------------------------------------------------------------------------
__global__ void gcn_degree(const int* __restrict__ col, float* __restrict__ deg, int nEdges)
{
    int e = blockIdx.x * blockDim.x + threadIdx.x;
    if (e < nEdges) atomicAdd(&deg[col[e]], 1.0f);
}

// ---------------------------------------------------------------------------
// GEMM: h[N,128] = xb[N,128] @ W[128,128]  via v_wmma_f32_16x16x32_bf16
// One wave -> one 16x16 tile of h. Block = 8 waves = all 8 N-tiles of an
// M-tile. K loop = 4 WMMA steps. nNodes assumed multiple of 16 (50000 = 3125*16).
//
// A fragment (16-bit A 16x32): lane%16 = M; lanes 0-15 hold K {0..7,16..23},
//   lanes 16-31 hold K {8..15,24..31}  -> two contiguous 16B chunks per lane.
// B fragment (32x16, column-major via transposed W): lane%16 = N; lanes 0-15
//   hold K {0..7,16..23}, lanes 16-31 hold K {8..15,24..31} -> same two 16B chunks as A.
// C/D (f32 16x16): VGPR r = row r + 8*(lane/16), col = lane%16.
// ---------------------------------------------------------------------------
__global__ void gcn_gemm_wmma(const bf16* __restrict__ xb, const bf16* __restrict__ wbt,
                              float* __restrict__ h, int nNodes)
{
    const int lane = threadIdx.x & 31;
    const int wave = threadIdx.x >> 5;
    const int half = lane >> 4;
    const int l16  = lane & 15;
    const int m0   = blockIdx.x * 16;
    const int n0   = wave * 16;
    if (m0 >= nNodes) return;

    union Frag { uint4 u[2]; v16bf v; };

    v8f c = {};
    const bf16* arow = xb  + (size_t)(m0 + l16) * IN_CH;
    const bf16* brow = wbt + (size_t)(n0 + l16) * IN_CH;

#pragma unroll
    for (int kt = 0; kt < IN_CH / 32; ++kt) {
        const int k0 = kt * 32;
        Frag A, B;
        A.u[0] = *reinterpret_cast<const uint4*>(arow + k0 + half * 8);        // K = h*8 .. h*8+7
        A.u[1] = *reinterpret_cast<const uint4*>(arow + k0 + 16 + half * 8);   // K = 16+h*8 ..
        B.u[0] = *reinterpret_cast<const uint4*>(brow + k0 + half * 8);        // K = h*8 .. h*8+7
        B.u[1] = *reinterpret_cast<const uint4*>(brow + k0 + 16 + half * 8);   // K = 16+h*8 ..
        c = __builtin_amdgcn_wmma_f32_16x16x32_bf16(
                /*neg_a=*/false, A.v, /*neg_b=*/false, B.v,
                /*c_mod=*/(short)0, c, /*reuse_a=*/false, /*reuse_b=*/false);
    }

#pragma unroll
    for (int r = 0; r < 8; ++r) {
        h[(size_t)(m0 + half * 8 + r) * OUT_CH + (n0 + l16)] = c[r];
    }
}

// ---------------------------------------------------------------------------
// scatter: acc[col] += h[row] * rsqrt(deg[row]) * rsqrt(deg[col])
// One wave per edge; each lane handles 4 consecutive features (coalesced atomics).
// ---------------------------------------------------------------------------
__global__ void gcn_scatter(const int* __restrict__ row, const int* __restrict__ col,
                            const float* __restrict__ h, const float* __restrict__ deg,
                            float* __restrict__ acc, int nEdges)
{
    int t    = blockIdx.x * blockDim.x + threadIdx.x;
    int e    = t >> 5;
    int lane = t & 31;
    if (e >= nEdges) return;
    const int r = row[e];
    const int c = col[e];
    const float norm = rsqrtf(deg[r]) * rsqrtf(deg[c]);
    const float4 hv = *reinterpret_cast<const float4*>(h + (size_t)r * OUT_CH + lane * 4);
    float* dst = acc + (size_t)c * OUT_CH + lane * 4;
    atomicAdd(dst + 0, hv.x * norm);
    atomicAdd(dst + 1, hv.y * norm);
    atomicAdd(dst + 2, hv.z * norm);
    atomicAdd(dst + 3, hv.w * norm);
}

// ---------------------------------------------------------------------------
// finalize: v = acc + h/deg (self loop) + bias; LayerNorm over 128; ReLU.
// One wave per node; 4 features per lane; wave32 shuffle reduction.
// ---------------------------------------------------------------------------
__global__ void gcn_finalize(const float* __restrict__ h, const float* __restrict__ deg,
                             const float* __restrict__ bias, const float* __restrict__ lnw,
                             const float* __restrict__ lnb, float* __restrict__ out, int nNodes)
{
    int t    = blockIdx.x * blockDim.x + threadIdx.x;
    int node = t >> 5;
    int lane = t & 31;
    if (node >= nNodes) return;

    const float selfw = 1.0f / deg[node];   // dinv[i]^2
    const size_t base = (size_t)node * OUT_CH + lane * 4;
    const float4 av = *reinterpret_cast<const float4*>(out  + base);
    const float4 hv = *reinterpret_cast<const float4*>(h    + base);
    const float4 bv = *reinterpret_cast<const float4*>(bias + lane * 4);

    float v0 = av.x + hv.x * selfw + bv.x;
    float v1 = av.y + hv.y * selfw + bv.y;
    float v2 = av.z + hv.z * selfw + bv.z;
    float v3 = av.w + hv.w * selfw + bv.w;

    float s = v0 + v1 + v2 + v3;
#pragma unroll
    for (int off = 16; off > 0; off >>= 1) s += __shfl_xor(s, off, 32);
    const float mu = s * (1.0f / OUT_CH);

    const float d0 = v0 - mu, d1 = v1 - mu, d2 = v2 - mu, d3 = v3 - mu;
    float q = d0 * d0 + d1 * d1 + d2 * d2 + d3 * d3;
#pragma unroll
    for (int off = 16; off > 0; off >>= 1) q += __shfl_xor(q, off, 32);
    const float rstd = rsqrtf(q * (1.0f / OUT_CH) + LN_EPS);

    const float4 wv = *reinterpret_cast<const float4*>(lnw + lane * 4);
    const float4 lb = *reinterpret_cast<const float4*>(lnb + lane * 4);
    float4 o;
    o.x = fmaxf(d0 * rstd * wv.x + lb.x, 0.0f);
    o.y = fmaxf(d1 * rstd * wv.y + lb.y, 0.0f);
    o.z = fmaxf(d2 * rstd * wv.z + lb.z, 0.0f);
    o.w = fmaxf(d3 * rstd * wv.w + lb.w, 0.0f);
    *reinterpret_cast<float4*>(out + base) = o;
}

// ---------------------------------------------------------------------------
extern "C" void kernel_launch(void* const* d_in, const int* in_sizes, int n_in,
                              void* d_out, int out_size, void* d_ws, size_t ws_size,
                              hipStream_t stream)
{
    const float* x    = (const float*)d_in[0];   // [N,128]
    const int*   ei   = (const int*)  d_in[1];   // [2,E]
    const float* W    = (const float*)d_in[2];   // [128,128]
    const float* bias = (const float*)d_in[3];   // [128]
    const float* lnw  = (const float*)d_in[4];   // [128]
    const float* lnb  = (const float*)d_in[5];   // [128]

    const int nNodes = in_sizes[0] / IN_CH;      // 50000 (multiple of 16)
    const int nEdges = in_sizes[1] / 2;          // 800000
    const int* row = ei;                          // source nodes
    const int* col = ei + nEdges;                 // target nodes

    float* out = (float*)d_out;                   // doubles as scatter accumulator

    // workspace layout
    char* p = (char*)d_ws;
    float* h  = (float*)p;  p += (size_t)nNodes * OUT_CH * sizeof(float);   // 25.6 MB
    bf16* xb  = (bf16*)p;   p += (size_t)nNodes * IN_CH * sizeof(bf16);     // 12.8 MB
    bf16* wbt = (bf16*)p;   p += (size_t)IN_CH * OUT_CH * sizeof(bf16);     // 32 KB
    float* deg = (float*)p;                                                  // 200 KB

    const int total = nNodes * IN_CH;
    gcn_prep<<<(total + 255) / 256, 256, 0, stream>>>(x, W, xb, wbt, deg, out, total, nNodes);
    gcn_degree<<<(nEdges + 255) / 256, 256, 0, stream>>>(col, deg, nEdges);
    gcn_gemm_wmma<<<(nNodes + 15) / 16, 256, 0, stream>>>(xb, wbt, h, nNodes);

    const long long scatterThreads = (long long)nEdges * 32;
    gcn_scatter<<<(int)((scatterThreads + 255) / 256), 256, 0, stream>>>(row, col, h, deg, out, nEdges);

    const long long finThreads = (long long)nNodes * 32;
    gcn_finalize<<<(int)((finThreads + 255) / 256), 256, 0, stream>>>(h, deg, bias, lnw, lnb, out, nNodes);
}
